// IntegralLNNLayer_120259085015
// MI455X (gfx1250) — compile-verified, matrix operand order checked by reading, not run on hardware
//
#include <hip/hip_runtime.h>
#include <hip/hip_bf16.h>

// ---------------------------------------------------------------------------
// Problem constants (from reference)
// ---------------------------------------------------------------------------
#define BB 4
#define SS 4096
#define DM 1024
#define KK 7
#define LCHUNK 256            // scan chunk length (power of 2)
#define NCHUNK (SS / LCHUNK)  // 16
#define RMS_EPS 1.1920929e-07f

// LDS tile row stride (bf16 elements): 40*2B = 80B, multiple of 16B so B128
// accesses stay aligned; 20-dword stride => conflict-minimal wave32 ds reads.
#define LSTRIDE 40
#define NBUF 4                // async pipeline ring buffers
#define PFD  3                // prefetch distance (tiles in flight)

typedef __attribute__((ext_vector_type(16))) __bf16 v16bf;
typedef __attribute__((ext_vector_type(8)))  __bf16 v8bf;
typedef __attribute__((ext_vector_type(4)))  __bf16 v4bf;
typedef __attribute__((ext_vector_type(8)))  float  v8f;

// ---------------------------------------------------------------------------
// CDNA5 async global->LDS copy (ASYNCcnt-tracked, no VGPR staging).
// 16 bytes per lane per instruction (512B per wave32 issue).
// ---------------------------------------------------------------------------
__device__ __forceinline__ void async_copy_b128(__bf16* lds_dst,
                                                const __bf16* gsrc) {
    unsigned loff =
        (unsigned)(uintptr_t)(__attribute__((address_space(3))) __bf16*)lds_dst;
    asm volatile("global_load_async_to_lds_b128 %0, %1, off"
                 :: "v"(loff), "v"(gsrc) : "memory");
}

// Wait until this wave's outstanding async ops <= N (loads retire in order,
// so N = 3*(PFD-1) guarantees the oldest tile's 3 fills are complete).
template <int N>
__device__ __forceinline__ void wait_async_le() {
#if __has_builtin(__builtin_amdgcn_s_wait_asynccnt)
    __builtin_amdgcn_s_wait_asynccnt(N);
#else
    asm volatile("s_wait_asynccnt %0" :: "i"(N) : "memory");
#endif
}

// ---------------------------------------------------------------------------
// Kernel 1: depthwise conv (K=7, pad 3) + sigmoid*tanh gating + chunk-local
// decay scan.  One thread per (b, d, chunk); loops 256 sequential steps.
// ---------------------------------------------------------------------------
__global__ void conv_gate_scan_kernel(const float* __restrict__ x,
                                      const float* __restrict__ gate_w,
                                      const float* __restrict__ gate_b,
                                      const float* __restrict__ cand_w,
                                      const float* __restrict__ cand_b,
                                      const float* __restrict__ log_decay,
                                      float* __restrict__ h,
                                      float* __restrict__ carry) {
    const int d = blockIdx.x * blockDim.x + threadIdx.x;  // 0..1023
    const int c = blockIdx.y;                             // chunk 0..15
    const int b = blockIdx.z;                             // batch 0..3
    const int s0 = c * LCHUNK;

    float gw[KK], cw[KK];
#pragma unroll
    for (int k = 0; k < KK; ++k) {
        gw[k] = gate_w[d * KK + k];
        cw[k] = cand_w[d * KK + k];
    }
    const float gb = gate_b[d];
    const float cb = cand_b[d];
    const float decay = __expf(log_decay[d]);

    const float* xb = x + (size_t)b * SS * DM + d;

    float win[KK];
#pragma unroll
    for (int j = 1; j < KK; ++j) {
        const int sg = s0 - 4 + j;
        win[j] = (sg >= 0 && sg < SS) ? xb[(size_t)sg * DM] : 0.0f;
    }

    float hs = 0.0f;  // chunk-local scan
    float* hb = h + (size_t)b * SS * DM + d;

    for (int s = s0; s < s0 + LCHUNK; ++s) {
#pragma unroll
        for (int j = 0; j < KK - 1; ++j) win[j] = win[j + 1];
        const int sg = s + 3;
        win[KK - 1] = (sg < SS) ? xb[(size_t)sg * DM] : 0.0f;

        float ga = gb, ca = cb;
#pragma unroll
        for (int k = 0; k < KK; ++k) {
            ga = fmaf(win[k], gw[k], ga);
            ca = fmaf(win[k], cw[k], ca);
        }
        const float gate = 1.0f / (1.0f + __expf(-ga));
        const float cand = tanhf(ca);
        hs = fmaf(decay, hs, gate * cand);
        hb[(size_t)s * DM] = hs;
    }
    carry[((size_t)b * NCHUNK + c) * DM + d] = hs;
}

// ---------------------------------------------------------------------------
// Kernel 2: combine chunk carries: prefix[c] = true h at end of chunk c-1.
// ---------------------------------------------------------------------------
__global__ void chunk_prefix_kernel(const float* __restrict__ carry,
                                    const float* __restrict__ log_decay,
                                    float* __restrict__ prefix) {
    const int idx = blockIdx.x * blockDim.x + threadIdx.x;  // 0..4095
    const int b = idx >> 10;
    const int d = idx & (DM - 1);
    const float decayL = __expf(log_decay[d] * (float)LCHUNK);
    float P = 0.0f;
#pragma unroll
    for (int c = 0; c < NCHUNK; ++c) {
        const size_t o = ((size_t)b * NCHUNK + c) * DM + d;
        prefix[o] = P;
        P = fmaf(decayL, P, carry[o]);
    }
}

// ---------------------------------------------------------------------------
// Kernel 3: cross-chunk fixup h[s] += decay^(slocal+1) * prefix -> bf16.
// ---------------------------------------------------------------------------
__global__ void fixup_bf16_kernel(const float* __restrict__ h,
                                  const float* __restrict__ prefix,
                                  const float* __restrict__ log_decay,
                                  __bf16* __restrict__ hbf) {
    const size_t e4 = ((size_t)blockIdx.x * blockDim.x + threadIdx.x) * 4;
    const int d0 = (int)(e4 & (DM - 1));
    const int row = (int)(e4 >> 10);       // b*S + s
    const int s = row & (SS - 1);
    const int b = row >> 12;
    const int chunk = s >> 8;
    const float sl1 = (float)((s & (LCHUNK - 1)) + 1);

    const float4 hv = *(const float4*)(h + e4);
    const size_t po = ((size_t)b * NCHUNK + chunk) * DM + d0;
    float r[4] = {hv.x, hv.y, hv.z, hv.w};
    v4bf o;
#pragma unroll
    for (int j = 0; j < 4; ++j) {
        const float f = __expf(log_decay[d0 + j] * sl1);
        o[j] = (__bf16)fmaf(f, prefix[po + j], r[j]);
    }
    *(v4bf*)(hbf + e4) = o;
}

// ---------------------------------------------------------------------------
// Kernel 4: convert out_w (D x D) to bf16.
// ---------------------------------------------------------------------------
__global__ void wconv_bf16_kernel(const float* __restrict__ w,
                                  __bf16* __restrict__ wbf) {
    const size_t e4 = ((size_t)blockIdx.x * blockDim.x + threadIdx.x) * 4;
    const float4 v = *(const float4*)(w + e4);
    v4bf o;
    o[0] = (__bf16)v.x; o[1] = (__bf16)v.y;
    o[2] = (__bf16)v.z; o[3] = (__bf16)v.w;
    *(v4bf*)(wbf + e4) = o;
}

// ---------------------------------------------------------------------------
// Kernel 5: GEMM y[m,e] = sum_d h[m,d]*W[e,d] + out_b[e] + x[m,e] via
// v_wmma_f32_16x16x32_bf16.  Block 256 thr = 8 waves, tile 64(M) x 128(N).
// 4-deep LDS ring filled by global_load_async_to_lds_b128: 3 K-slices of HBM
// traffic stay in flight behind the WMMAs of the current slice.
// ---------------------------------------------------------------------------
__global__ void __launch_bounds__(256)
gemm_wmma_kernel(const __bf16* __restrict__ hbf,
                 const __bf16* __restrict__ wbf,
                 const float* __restrict__ out_b,
                 const float* __restrict__ x,
                 float* __restrict__ y) {
    __shared__ __align__(16) __bf16 As[NBUF][64 * LSTRIDE];    // 4 x 5 KB
    __shared__ __align__(16) __bf16 Bs[NBUF][128 * LSTRIDE];   // 4 x 10 KB

    const int tid = threadIdx.x;
    const int lane = tid & 31;
    const int wid = tid >> 5;          // 0..7
    const int wave_m = wid >> 1;       // 0..3
    const int wave_n = wid & 1;        // 0..1
    const int r = lane & 15;
    const int g = lane >> 4;

    const int rowblk = blockIdx.y * 64;
    const int colblk = blockIdx.x * 128;

    // Cooperative-load assignments (16B chunks of a 32-wide K slice).
    const int a_row = tid >> 2;            // 0..63
    const int a_cc  = (tid & 3) * 8;       // k sub-offset 0/8/16/24
    const int b_row0 = tid >> 2;           // 0..63
    const int b_row1 = (tid + 256) >> 2;   // 64..127
    const int b_cc  = (tid & 3) * 8;

    const __bf16* gA  = hbf + (size_t)(rowblk + a_row) * DM + a_cc;
    const __bf16* gB0 = wbf + (size_t)(colblk + b_row0) * DM + b_cc;
    const __bf16* gB1 = wbf + (size_t)(colblk + b_row1) * DM + b_cc;

    v8f acc0 = {0,0,0,0,0,0,0,0};
    v8f acc1 = {0,0,0,0,0,0,0,0};
    v8f acc2 = {0,0,0,0,0,0,0,0};
    v8f acc3 = {0,0,0,0,0,0,0,0};

    // Prologue: put PFD K-slices in flight.
#pragma unroll
    for (int p = 0; p < PFD; ++p) {
        async_copy_b128(&As[p][a_row * LSTRIDE + a_cc], gA + p * 32);
        async_copy_b128(&Bs[p][b_row0 * LSTRIDE + b_cc], gB0 + p * 32);
        async_copy_b128(&Bs[p][b_row1 * LSTRIDE + b_cc], gB1 + p * 32);
    }

    for (int kk = 0; kk < DM; kk += 32) {
        const int cur = (kk >> 5) & (NBUF - 1);
        // Oldest tile's fills complete (3*(PFD-1)=6 younger may remain).
        wait_async_le<3 * (PFD - 1)>();
        __syncthreads();       // all waves' fills of buf[cur] visible

        // Keep exactly PFD tiles in flight; tail issues clamp to the last
        // K-slice and refill dead buffers (uniform wait immediate, safe).
        {
            const int nxt = (cur + PFD) & (NBUF - 1);
            const int kn = (kk + PFD * 32 < DM) ? (kk + PFD * 32) : (DM - 32);
            async_copy_b128(&As[nxt][a_row * LSTRIDE + a_cc], gA + kn);
            async_copy_b128(&Bs[nxt][b_row0 * LSTRIDE + b_cc], gB0 + kn);
            async_copy_b128(&Bs[nxt][b_row1 * LSTRIDE + b_cc], gB1 + kn);
        }

        // Preload A fragment and all 4 B fragments, then WMMA back-to-back.
        const __bf16* la = &As[cur][(wave_m * 16 + r) * LSTRIDE + g * 8];
        const v8bf a_lo = *(const v8bf*)la;
        const v8bf a_hi = *(const v8bf*)(la + 16);
        const v16bf a = __builtin_shufflevector(a_lo, a_hi,
            0,1,2,3,4,5,6,7,8,9,10,11,12,13,14,15);

        v16bf bf[4];
#pragma unroll
        for (int t = 0; t < 4; ++t) {
            const __bf16* lb =
                &Bs[cur][(wave_n * 64 + t * 16 + r) * LSTRIDE + g * 8];
            const v8bf b_lo = *(const v8bf*)lb;
            const v8bf b_hi = *(const v8bf*)(lb + 16);
            bf[t] = __builtin_shufflevector(b_lo, b_hi,
                0,1,2,3,4,5,6,7,8,9,10,11,12,13,14,15);
        }
        acc0 = __builtin_amdgcn_wmma_f32_16x16x32_bf16(
                   false, a, false, bf[0], (short)0, acc0, false, false);
        acc1 = __builtin_amdgcn_wmma_f32_16x16x32_bf16(
                   false, a, false, bf[1], (short)0, acc1, false, false);
        acc2 = __builtin_amdgcn_wmma_f32_16x16x32_bf16(
                   false, a, false, bf[2], (short)0, acc2, false, false);
        acc3 = __builtin_amdgcn_wmma_f32_16x16x32_bf16(
                   false, a, false, bf[3], (short)0, acc3, false, false);
    }

    // Epilogue: (acc[v], lane) = out[M = v + 8*g][N = r].
    const int row0 = rowblk + wave_m * 16;
    const int col0 = colblk + wave_n * 64;
    v8f accs[4] = {acc0, acc1, acc2, acc3};
#pragma unroll
    for (int t = 0; t < 4; ++t) {
        const int col = col0 + t * 16 + r;
        const float bias = out_b[col];
#pragma unroll
        for (int v = 0; v < 8; ++v) {
            const int m = row0 + g * 8 + v;
            const size_t o = (size_t)m * DM + col;
            y[o] = accs[t][v] + bias + x[o];
        }
    }
}

// ---------------------------------------------------------------------------
// Kernel 6: RMSNorm over D=1024 per row.
// ---------------------------------------------------------------------------
__global__ void __launch_bounds__(256)
rmsnorm_kernel(const float* __restrict__ y,
               const float* __restrict__ norm_w,
               float* __restrict__ out) {
    __shared__ float red[256];
    const int row = blockIdx.x;
    const int tid = threadIdx.x;
    const float4 v = *(const float4*)(y + (size_t)row * DM + tid * 4);
    float ss = v.x * v.x + v.y * v.y + v.z * v.z + v.w * v.w;
    red[tid] = ss;
    __syncthreads();
#pragma unroll
    for (int off = 128; off > 0; off >>= 1) {
        if (tid < off) red[tid] += red[tid + off];
        __syncthreads();
    }
    const float rs = rsqrtf(red[0] * (1.0f / (float)DM) + RMS_EPS);
    const float4 nw = *(const float4*)(norm_w + tid * 4);
    float4 o;
    o.x = v.x * rs * nw.x;
    o.y = v.y * rs * nw.y;
    o.z = v.z * rs * nw.z;
    o.w = v.w * rs * nw.w;
    *(float4*)(out + (size_t)row * DM + tid * 4) = o;
}

// ---------------------------------------------------------------------------
// Host-side launcher
// ---------------------------------------------------------------------------
extern "C" void kernel_launch(void* const* d_in, const int* in_sizes, int n_in,
                              void* d_out, int out_size, void* d_ws, size_t ws_size,
                              hipStream_t stream) {
    const float* x         = (const float*)d_in[0];
    const float* gate_w    = (const float*)d_in[1];
    const float* gate_b    = (const float*)d_in[2];
    const float* cand_w    = (const float*)d_in[3];
    const float* cand_b    = (const float*)d_in[4];
    const float* log_decay = (const float*)d_in[5];
    const float* out_w     = (const float*)d_in[6];
    const float* out_b     = (const float*)d_in[7];
    const float* norm_w    = (const float*)d_in[8];
    float* out = (float*)d_out;

    constexpr size_t H_F32  = (size_t)BB * SS * DM * 4;        // 64 MB
    constexpr size_t H_BF16 = (size_t)BB * SS * DM * 2;        // 32 MB
    constexpr size_t W_BF16 = (size_t)DM * DM * 2;             // 2 MB
    constexpr size_t CARRY  = (size_t)BB * DM * NCHUNK * 4;    // 256 KB

    char* ws = (char*)d_ws;
    float*  h_f32  = (float*)(ws);
    float*  y_f32  = (float*)(ws);                      // alias (h dead by then)
    __bf16* h_bf16 = (__bf16*)(ws + H_F32);
    __bf16* w_bf16 = (__bf16*)(ws + H_F32 + H_BF16);
    float*  carry  = (float*)(ws + H_F32 + H_BF16 + W_BF16);
    float*  prefix = (float*)(ws + H_F32 + H_BF16 + W_BF16 + CARRY);
    (void)ws_size; (void)in_sizes; (void)n_in; (void)out_size;

    conv_gate_scan_kernel<<<dim3(DM / 256, NCHUNK, BB), 256, 0, stream>>>(
        x, gate_w, gate_b, cand_w, cand_b, log_decay, h_f32, carry);

    chunk_prefix_kernel<<<(BB * DM) / 256, 256, 0, stream>>>(
        carry, log_decay, prefix);

    fixup_bf16_kernel<<<((size_t)BB * SS * DM / 4) / 256, 256, 0, stream>>>(
        h_f32, prefix, log_decay, h_bf16);

    wconv_bf16_kernel<<<((size_t)DM * DM / 4) / 256, 256, 0, stream>>>(
        out_w, w_bf16);

    gemm_wmma_kernel<<<dim3(DM / 128, (BB * SS) / 64), 256, 0, stream>>>(
        h_bf16, w_bf16, out_b, x, y_f32);

    rmsnorm_kernel<<<BB * SS, 256, 0, stream>>>(y_f32, norm_w, out);
}
